// CausalSelfAttention_27702539059819
// MI455X (gfx1250) — matrix operand order repaired and checked by reading, not compiled, not run on hardware
//
#include <hip/hip_runtime.h>
#include <cstdint>

// ---------------------------------------------------------------------------
// CausalSelfAttention forward on gfx1250 (CDNA5).
//   - f32 inputs converted once to f16; all matmuls V_WMMA_F32_16X16X32_F16
//     (f32 accumulate).  Working set (~50 MB) lives in the 192 MB L2, so the
//     kernel is WMMA-throughput bound; data movement uses the CDNA5 async
//     path (GLOBAL_LOAD_ASYNC_TO_LDS_B128 / ASYNCcnt) with double-buffered
//     LDS tiles so copies for tile t+1 overlap the WMMAs of tile t.
// ---------------------------------------------------------------------------

typedef __attribute__((ext_vector_type(16))) _Float16     v16h;
typedef __attribute__((ext_vector_type(8)))  float        v8f;
typedef __attribute__((ext_vector_type(4)))  unsigned int v4u;

__device__ __forceinline__ v8f wmma_f16(v16h a, v16h b, v8f c) {
  return __builtin_amdgcn_wmma_f32_16x16x32_f16(false, a, false, b, (short)0, c,
                                                false, false);
}

// Low 32 bits of a generic pointer to __shared__ = LDS byte address.
__device__ __forceinline__ unsigned lds_addr(const void* p) {
  return (unsigned)(uintptr_t)p;
}

// One lane copies 16 bytes global->LDS, tracked by ASYNCcnt (GVS addressing).
__device__ __forceinline__ void async_b128(unsigned dst_lds, unsigned src_off,
                                           const void* base) {
  asm volatile("global_load_async_to_lds_b128 %0, %1, %2"
               :: "v"(dst_lds), "v"(src_off), "s"(base) : "memory");
}
__device__ __forceinline__ void async_wait() {
  asm volatile("s_wait_asynccnt 0x0" ::: "memory");
}

// A fragment (16x32 f16): lane L -> row M = L%16; element e ->
// K = e + 8*(e>=8) + 8*(L>=16).  Pairs contiguous -> b32 reads (ld even).
__device__ __forceinline__ v16h load_a_frag(const _Float16* p, int ld,
                                            int rowbase, int kbase, int lane) {
  union { v16h v; unsigned int u[8]; } f;
  const int hi  = (lane >> 4) & 1;
  const int row = rowbase + (lane & 15);
  const _Float16* rp = p + row * ld + kbase + hi * 8;
#pragma unroll
  for (int e2 = 0; e2 < 8; ++e2) {
    const int K = 2 * e2 + ((e2 >= 4) ? 8 : 0);
    f.u[e2] = *reinterpret_cast<const unsigned int*>(rp + K);
  }
  return f.v;
}

// B fragment (32x16 f16): lane L -> col N = L%16; element e ->
// K = e + 8*(e>=8) + 8*(L>=16).  Scalar half reads.
__device__ __forceinline__ v16h load_b_frag(const _Float16* p, int ld,
                                            int kbase, int nbase, int lane) {
  union { v16h v; _Float16 h[16]; } f;
  const int hi = (lane >> 4) & 1;
  const _Float16* cp = p + (kbase + hi * 8) * ld + nbase + (lane & 15);
#pragma unroll
  for (int e = 0; e < 16; ++e) {
    const int K = e + ((e >= 8) ? 8 : 0);
    f.h[e] = cp[K * ld];
  }
  return f.v;
}

// ---------------------------------------------------------------------------
// f32 -> f16 one-time conversion (vectorized x4).
// ---------------------------------------------------------------------------
__global__ __launch_bounds__(256) void cvt_f32_to_f16(
    const float* __restrict__ in, _Float16* __restrict__ out, int n4) {
  const int i = blockIdx.x * 256 + threadIdx.x;
  if (i >= n4) return;
  const float4 v = reinterpret_cast<const float4*>(in)[i];
  union { _Float16 h[4]; uint2 u; } o;
  o.h[0] = (_Float16)v.x; o.h[1] = (_Float16)v.y;
  o.h[2] = (_Float16)v.z; o.h[3] = (_Float16)v.w;
  reinterpret_cast<uint2*>(out)[i] = o.u;
}

// ---------------------------------------------------------------------------
// GEMM + bias: Cout[M,N] = A[M,K] * W[K,N] + bias[N]   (f16 in, f32 accum)
// 256 threads = 8 waves; block tile 128x128, BK=32; wave tile 32x64
// (2 A-frags x 4 B-frags -> 8 WMMAs per K-step).  Double-buffered async LDS.
// ---------------------------------------------------------------------------
template <typename OutT>
__global__ __launch_bounds__(256) void gemm_bias_f16(
    const _Float16* __restrict__ A, const _Float16* __restrict__ W,
    const float* __restrict__ bias, OutT* __restrict__ Cout,
    int M, int N, int K) {
  constexpr int ldA = 40;   // 32+8 pad halfs (80B rows, 16B chunk aligned)
  constexpr int ldB = 136;  // 128+8 pad halfs (272B rows, 16B chunk aligned)
  __shared__ __align__(16) _Float16 Alds[2][128 * ldA];
  __shared__ __align__(16) _Float16 Blds[2][32 * ldB];

  const int tid  = threadIdx.x;
  const int lane = tid & 31;
  const int wave = tid >> 5;
  const int rw   = wave & 3;   // 32-row strip
  const int cw   = wave >> 2;  // 64-col strip
  const int mbase = blockIdx.y * 128;
  const int nbase = blockIdx.x * 128;

  // A tile: 128 rows x 4 chunks of 8 halfs = 512 chunks -> 2 per thread.
  const int ar = tid >> 2, ac = (tid & 3) * 8;
  // B tile: 32 rows x 16 chunks = 512 chunks -> 2 per thread.
  const int br = tid >> 4, bc = (tid & 15) * 8;

  auto stage = [&](int kt, int ib) {
    const unsigned aB = lds_addr(&Alds[ib][0]);
    const unsigned bB = lds_addr(&Blds[ib][0]);
    async_b128(aB + (unsigned)(ar * ldA + ac) * 2,
               (unsigned)(((mbase + ar) * K + kt + ac) * 2), A);
    async_b128(aB + (unsigned)((ar + 64) * ldA + ac) * 2,
               (unsigned)(((mbase + ar + 64) * K + kt + ac) * 2), A);
    async_b128(bB + (unsigned)(br * ldB + bc) * 2,
               (unsigned)(((kt + br) * N + nbase + bc) * 2), W);
    async_b128(bB + (unsigned)((br + 16) * ldB + bc) * 2,
               (unsigned)(((kt + br + 16) * N + nbase + bc) * 2), W);
  };

  v8f acc[2][4];
#pragma unroll
  for (int i = 0; i < 2; ++i)
#pragma unroll
    for (int j = 0; j < 4; ++j) acc[i][j] = (v8f){};

  stage(0, 0);
  int ib = 0;
  for (int kt = 0; kt < K; kt += 32, ib ^= 1) {
    async_wait();        // own copies for current buffer done
    __syncthreads();     // -> every wave's copies done; prior reads retired
    if (kt + 32 < K) stage(kt + 32, ib ^ 1);  // overlap with compute below

    const v16h a0 = load_a_frag(&Alds[ib][0], ldA, rw * 32,      0, lane);
    const v16h a1 = load_a_frag(&Alds[ib][0], ldA, rw * 32 + 16, 0, lane);
#pragma unroll
    for (int ns = 0; ns < 4; ++ns) {
      const v16h b = load_b_frag(&Blds[ib][0], ldB, 0, cw * 64 + ns * 16, lane);
      acc[0][ns] = wmma_f16(a0, b, acc[0][ns]);
      acc[1][ns] = wmma_f16(a1, b, acc[1][ns]);
    }
    __syncthreads();     // reads of this buffer done before it is re-filled
  }

  const int hi = (lane >> 4) & 1;
  const int nc = lane & 15;
#pragma unroll
  for (int ns = 0; ns < 4; ++ns) {
    const int col = nbase + cw * 64 + ns * 16 + nc;
    const float bf = bias[col];
#pragma unroll
    for (int r = 0; r < 8; ++r) {
      const int row = mbase + rw * 32 + hi * 8 + r;
      Cout[(size_t)row * N + col]        = (OutT)(acc[0][ns][r] + bf);
      Cout[(size_t)(row + 16) * N + col] = (OutT)(acc[1][ns][r] + bf);
    }
  }
}

// ---------------------------------------------------------------------------
// Flash attention: one block = (b, h, 64-query tile), 128 threads = 4 waves.
// Double-buffered K/V staging: V via async b128, K via early global loads +
// late LDS transpose scatter.  P overlays the S buffer (f16 pairs packed and
// stored as f32 into the same array, so the dependence is type-visible).
// ---------------------------------------------------------------------------
__global__ __launch_bounds__(128) void attn_kernel(
    const _Float16* __restrict__ qkv, _Float16* __restrict__ y) {
  constexpr int T = 2048, Cc = 1024, D = 64;
  constexpr int C3 = 3 * Cc;
  constexpr int ldQ = 72, ldK = 72, ldV = 72, ldS = 65;
  constexpr int ldP = 2 * ldS;  // P half-stride inside the S buffer

  __shared__ __align__(16) _Float16 Qs [64 * ldQ];      // [qrow][d]
  __shared__ __align__(16) _Float16 Kts[2][64 * ldK];   // [d][key]
  __shared__ __align__(16) _Float16 Vs [2][64 * ldV];   // [key][d]
  __shared__ __align__(16) float    Ss [64 * ldS];      // scores f32 / P f16
  __shared__ float scaleArr[64];
  __shared__ float lArr[64];

  const int tid  = threadIdx.x;
  const int lane = tid & 31;
  const int wave = tid >> 5;
  const int rowbase = wave * 16;
  const int hi = (lane >> 4) & 1;
  const int nc = lane & 15;

  const int wg = blockIdx.x;      // B*H*(T/64) = 1024
  const int qb = wg & 31;
  const int h  = (wg >> 5) & 15;
  const int bb = wg >> 9;

  // Per-thread chunk coords: 64 rows x 8 chunks of 8 halfs, 4 per thread.
  v4u kreg[4];

  auto stage_v = [&](int j, int ib) {
    const unsigned vB = lds_addr(&Vs[ib][0]);
#pragma unroll
    for (int i = 0; i < 4; ++i) {
      const int g = tid + i * 128;
      const int r = g >> 3, cc = (g & 7) * 8;
      async_b128(vB + (unsigned)(r * ldV + cc) * 2,
                 (unsigned)(((bb * T + j * 64 + r) * C3 + 2 * Cc + h * D + cc) * 2),
                 qkv);
    }
  };
  auto load_k = [&](int j) {
#pragma unroll
    for (int i = 0; i < 4; ++i) {
      const int g = tid + i * 128;
      const int key = g >> 3, cc = (g & 7) * 8;
      kreg[i] = *reinterpret_cast<const v4u*>(
          qkv + (size_t)(bb * T + j * 64 + key) * C3 + Cc + h * D + cc);
    }
  };
  auto store_k = [&](int ib) {
#pragma unroll
    for (int i = 0; i < 4; ++i) {
      const int g = tid + i * 128;
      const int key = g >> 3, cc = (g & 7) * 8;
      union { v4u q; _Float16 hh[8]; } t;
      t.q = kreg[i];
#pragma unroll
      for (int dd = 0; dd < 8; ++dd) Kts[ib][(cc + dd) * ldK + key] = t.hh[dd];
    }
  };

  // Prologue: stage Q, V(0) (async) and K(0) (regs -> LDS transpose).
  {
    const unsigned qB = lds_addr(Qs);
#pragma unroll
    for (int i = 0; i < 4; ++i) {
      const int g = tid + i * 128;
      const int r = g >> 3, cc = (g & 7) * 8;
      async_b128(qB + (unsigned)(r * ldQ + cc) * 2,
                 (unsigned)(((bb * T + qb * 64 + r) * C3 + h * D + cc) * 2), qkv);
    }
  }
  stage_v(0, 0);
  load_k(0);
  store_k(0);
  async_wait();
  __syncthreads();

  const v16h aq0 = load_a_frag(Qs, ldQ, rowbase, 0, lane);
  const v16h aq1 = load_a_frag(Qs, ldQ, rowbase, 32, lane);

  v8f o0 = {}, o1 = {}, o2 = {}, o3 = {};
  float m = -__builtin_inff();
  float l = 0.0f;

  for (int j = 0; j <= qb; ++j) {
    const int cur = j & 1, nxt = cur ^ 1;
    // Kick off next block's copies; they complete during this block's math.
    if (j < qb) { stage_v(j + 1, nxt); load_k(j + 1); }

    // S = Q K^T (scale 1/sqrt(64) folded into the f32 store).
#pragma unroll
    for (int ns = 0; ns < 4; ++ns) {
      v8f s = {};
      s = wmma_f16(aq0, load_b_frag(&Kts[cur][0], ldK, 0,  ns * 16, lane), s);
      s = wmma_f16(aq1, load_b_frag(&Kts[cur][0], ldK, 32, ns * 16, lane), s);
#pragma unroll
      for (int r = 0; r < 8; ++r)
        Ss[(rowbase + hi * 8 + r) * ldS + ns * 16 + nc] = s[r] * 0.125f;
    }
    __syncthreads();

    // Online softmax, one thread per query row.  P is written back into the
    // S buffer as packed f16 pairs via f32-typed stores; the write at pair
    // index i/2 only touches S elements already consumed (i/2 < i).
    if (tid < 64) {
      const int row    = tid;
      const int nvalid = (j == qb) ? (row + 1) : 64;
      float* srow = &Ss[row * ldS];
      float rmax = -__builtin_inff();
      for (int i2 = 0; i2 < nvalid; ++i2) rmax = fmaxf(rmax, srow[i2]);
      const float mnew  = fmaxf(m, rmax);
      const float alpha = __expf(m - mnew);
      float psum = 0.0f;
      for (int i2 = 0; i2 < 64; i2 += 2) {
        const float s0 = srow[i2], s1 = srow[i2 + 1];
        const float p0 = (i2     < nvalid) ? __expf(s0 - mnew) : 0.0f;
        const float p1 = (i2 + 1 < nvalid) ? __expf(s1 - mnew) : 0.0f;
        union { _Float16 hh[2]; float f; } pk;
        pk.hh[0] = (_Float16)p0; pk.hh[1] = (_Float16)p1;
        srow[i2 >> 1] = pk.f;
        psum += p0 + p1;
      }
      l = l * alpha + psum;
      m = mnew;
      scaleArr[row] = alpha;
    }
    __syncthreads();

    // Rescale accumulators per query row.
    float al[8];
#pragma unroll
    for (int r = 0; r < 8; ++r) al[r] = scaleArr[rowbase + hi * 8 + r];
#pragma unroll
    for (int r = 0; r < 8; ++r) {
      o0[r] *= al[r]; o1[r] *= al[r]; o2[r] *= al[r]; o3[r] *= al[r];
    }

    // O += P * V.
    const _Float16* Ps = reinterpret_cast<const _Float16*>(Ss);
    const v16h ap0 = load_a_frag(Ps, ldP, rowbase, 0, lane);
    const v16h ap1 = load_a_frag(Ps, ldP, rowbase, 32, lane);
    const _Float16* Vc = &Vs[cur][0];
    o0 = wmma_f16(ap0, load_b_frag(Vc, ldV, 0,  0,  lane), o0);
    o0 = wmma_f16(ap1, load_b_frag(Vc, ldV, 32, 0,  lane), o0);
    o1 = wmma_f16(ap0, load_b_frag(Vc, ldV, 0,  16, lane), o1);
    o1 = wmma_f16(ap1, load_b_frag(Vc, ldV, 32, 16, lane), o1);
    o2 = wmma_f16(ap0, load_b_frag(Vc, ldV, 0,  32, lane), o2);
    o2 = wmma_f16(ap1, load_b_frag(Vc, ldV, 32, 32, lane), o2);
    o3 = wmma_f16(ap0, load_b_frag(Vc, ldV, 0,  48, lane), o3);
    o3 = wmma_f16(ap1, load_b_frag(Vc, ldV, 32, 48, lane), o3);

    // Late LDS scatter of next K block; global loads had the whole compute
    // phase to land.  Then fence asyncs + barrier before the buffer swap.
    if (j < qb) { store_k(nxt); async_wait(); }
    __syncthreads();
  }

  if (tid < 64) lArr[tid] = l;
  __syncthreads();

  float inv[8];
#pragma unroll
  for (int r = 0; r < 8; ++r) inv[r] = 1.0f / lArr[rowbase + hi * 8 + r];

  const int trow0 = qb * 64 + rowbase + hi * 8;
#pragma unroll
  for (int r = 0; r < 8; ++r) {
    _Float16* yrow = y + (size_t)(bb * T + trow0 + r) * Cc + h * D + nc;
    yrow[0]  = (_Float16)(o0[r] * inv[r]);
    yrow[16] = (_Float16)(o1[r] * inv[r]);
    yrow[32] = (_Float16)(o2[r] * inv[r]);
    yrow[48] = (_Float16)(o3[r] * inv[r]);
  }
}

// ---------------------------------------------------------------------------
// Launch: convert -> qkv GEMM (f16 out) -> flash attention -> out projection.
// Workspace (halfs): xh 4M | w_attnh 3M | w_projh 1M | qkvh 12M | yh 4M = 48MB.
// ---------------------------------------------------------------------------
extern "C" void kernel_launch(void* const* d_in, const int* in_sizes, int n_in,
                              void* d_out, int out_size, void* d_ws, size_t ws_size,
                              hipStream_t stream) {
  (void)in_sizes; (void)n_in; (void)out_size; (void)ws_size;
  constexpr int B = 2, T = 2048, C = 1024;
  constexpr int M  = B * T;   // 4096
  constexpr int N3 = 3 * C;   // 3072

  const float* x      = (const float*)d_in[0];
  const float* w_attn = (const float*)d_in[1];
  const float* b_attn = (const float*)d_in[2];
  const float* w_proj = (const float*)d_in[3];
  const float* b_proj = (const float*)d_in[4];
  float* out = (float*)d_out;

  _Float16* xh      = (_Float16*)d_ws;
  _Float16* w_attnh = xh      + (size_t)M * C;
  _Float16* w_projh = w_attnh + (size_t)C * N3;
  _Float16* qkvh    = w_projh + (size_t)C * C;
  _Float16* yh      = qkvh    + (size_t)M * N3;

  cvt_f32_to_f16<<<(M * C / 4 + 255) / 256, 256, 0, stream>>>(x, xh, M * C / 4);
  cvt_f32_to_f16<<<(C * N3 / 4 + 255) / 256, 256, 0, stream>>>(w_attn, w_attnh,
                                                               C * N3 / 4);
  cvt_f32_to_f16<<<(C * C / 4 + 255) / 256, 256, 0, stream>>>(w_proj, w_projh,
                                                              C * C / 4);

  gemm_bias_f16<_Float16><<<dim3(N3 / 128, M / 128), 256, 0, stream>>>(
      xh, w_attnh, b_attn, qkvh, M, N3, C);

  attn_kernel<<<dim3(B * 16 * (T / 64)), 128, 0, stream>>>(qkvh, yh);

  gemm_bias_f16<float><<<dim3(C / 128, M / 128), 256, 0, stream>>>(
      yh, w_projh, b_proj, out, M, C, C);
}
